// GLORYSplit_47571057770693
// MI455X (gfx1250) — matrix operand order repaired and checked by reading, not compiled, not run on hardware
//
#include <hip/hip_runtime.h>
#include <math.h>

// ---------------- problem constants (match reference) ----------------
constexpr int NN    = 10000;   // graph nodes
constexpr int EE    = 200000;  // edges
constexpr int BB    = 64;      // batch
constexpr int NC    = 50;      // clicked slots
constexpr int NCAND = 5;
constexpr int TOK   = 38;
constexpr int TTL   = 30;      // title tokens used
constexpr int GD    = 300;     // glove dim
constexpr int DD    = 400;     // news dim
constexpr int DP    = 416;     // DD padded to multiple of 32
constexpr int HH    = 20;      // heads
constexpr int DH    = 20;      // head dim
constexpr float INVS = 0.22360679775f; // 1/sqrt(20)

// ---------------- WMMA plumbing (gfx1250, wave32) ----------------
typedef __attribute__((ext_vector_type(16))) _Float16 v16h;
typedef __attribute__((ext_vector_type(8)))  float    v8f;

union Frag16 { v16h v; uint4 q[2]; unsigned int u[8]; _Float16 e[16]; };
union HPair  { _Float16 h[2]; unsigned int u; };

__device__ __forceinline__ v8f wmma32(v16h a, v16h b, v8f c) {
  // emits v_wmma_f32_16x16x32_f16
  return __builtin_amdgcn_wmma_f32_16x16x32_f16(false, a, false, b, (short)0, c,
                                                false, false);
}

// A fragment (16x32) from padded row-major A[* x lda] (lda % 32 == 0, rows valid).
// ISA 7.12.2 16-bit A layout: lane L -> M=L%16 ; element pairs at
// k = k0 + (L/16)*8 + {0,2,4,6} and +16..  => two contiguous 16B runs.
__device__ __forceinline__ v16h load_a_p(const _Float16* A, int lda,
                                         int row0, int k0, int lane) {
  const _Float16* rp = A + (size_t)(row0 + (lane & 15)) * lda + k0 + ((lane >> 4) << 3);
  Frag16 f;
  f.q[0] = *(const uint4*)(rp);
  f.q[1] = *(const uint4*)(rp + 16);
  return f.v;
}

// B fragment from pre-packed weights: [nt][kb][lane][8 x u32], 32B per lane.
__device__ __forceinline__ v16h load_b_p(const unsigned int* Bp, int KB,
                                         int nt, int kb, int lane) {
  const uint4* p = (const uint4*)(Bp + ((((size_t)nt * KB + kb) << 5 | lane) << 3));
  Frag16 f;
  f.q[0] = p[0];
  f.q[1] = p[1];
  return f.v;
}

// ---------------- pack B[KxN] f32 -> WMMA fragment order f16 ----------------
// dst word idx = ((nt*KB + kb)*32 + lane)*8 + i holds (k,k+1) of col = nt*16+(lane&15)
// with k = kb*32 + (lane>>4)*8 + (i>>2)*16 + 2*(i&3) ; zero-filled outside [K,N].
__global__ void k_packB(const float* __restrict__ src, unsigned int* __restrict__ dst,
                        int K, int N, int KB, int NT) {
  int total = NT * KB * 256;
  for (int idx = blockIdx.x * blockDim.x + threadIdx.x; idx < total;
       idx += gridDim.x * blockDim.x) {
    int i    = idx & 7;
    int lane = (idx >> 3) & 31;
    int blk  = idx >> 8;
    int kb   = blk % KB, nt = blk / KB;
    int col  = nt * 16 + (lane & 15);
    int k    = kb * 32 + ((lane >> 4) << 3) + ((i >> 2) << 4) + 2 * (i & 3);
    HPair p;
    p.h[0] = (col < N && k     < K) ? (_Float16)src[(size_t)k       * N + col] : (_Float16)0.f;
    p.h[1] = (col < N && (k+1) < K) ? (_Float16)src[(size_t)(k + 1) * N + col] : (_Float16)0.f;
    dst[idx] = p.u;
  }
}

// ---------------- small utility kernels ----------------
__global__ void k_f32tof16_pad(const float* __restrict__ s, _Float16* __restrict__ d,
                               int M, int K, int Kp) {
  int total = M * Kp;
  for (int i = blockIdx.x * blockDim.x + threadIdx.x; i < total;
       i += gridDim.x * blockDim.x) {
    int r = i / Kp, c = i % Kp;
    d[i] = (c < K) ? (_Float16)s[(size_t)r * K + c] : (_Float16)0.f;
  }
}

__global__ void k_zero(float* __restrict__ d, int n) {
  for (int i = blockIdx.x * blockDim.x + threadIdx.x; i < n; i += gridDim.x * blockDim.x)
    d[i] = 0.f;
}

__global__ void k_zero16(_Float16* __restrict__ d, int n) {
  for (int i = blockIdx.x * blockDim.x + threadIdx.x; i < n; i += gridDim.x * blockDim.x)
    d[i] = (_Float16)0.f;
}

// ------------- generic WMMA GEMM: C = A[M x K(pad lda)] @ Bpacked (+bias) -------------
__global__ void __launch_bounds__(256)
k_gemm(const _Float16* __restrict__ A, int lda, const unsigned int* __restrict__ Bp,
       const float* __restrict__ bias, float* __restrict__ C,
       _Float16* __restrict__ C16, int ldc16, int M, int KB, int N) {
  int gw   = (blockIdx.x * 256 + threadIdx.x) >> 5;
  int lane = threadIdx.x & 31;
  int tm = (M + 15) >> 4, tn = (N + 15) >> 4;
  if (gw >= tm * tn) return;                 // wave-uniform exit
  int mt = gw / tn, nt = gw % tn;
  v8f c = {};
  for (int kb = 0; kb < KB; ++kb) {
    v16h a = load_a_p(A, lda, mt * 16, kb * 32, lane);
    v16h b = load_b_p(Bp, KB, nt, kb, lane);
    c = wmma32(a, b, c);
  }
  int col = nt * 16 + (lane & 15);
  int rbase = mt * 16 + ((lane >> 4) ? 8 : 0);
  if (col < N) {
    float bv = bias ? bias[col] : 0.f;
    #pragma unroll
    for (int r = 0; r < 8; ++r) {
      int row = rbase + r;
      float v = c[r] + bv;
      C[(size_t)row * N + col] = v;
      if (C16) C16[(size_t)row * ldc16 + col] = (_Float16)v;
    }
  }
}

// ---------------- news encoder: one block per title ----------------
__global__ void __launch_bounds__(256)
k_news(const int* __restrict__ subx, const int* __restrict__ cand,
       const int* __restrict__ fake, const float* __restrict__ glove,
       const unsigned int* __restrict__ wqP, const unsigned int* __restrict__ wkP,
       const unsigned int* __restrict__ wvP, const unsigned int* __restrict__ wpP,
       const float* __restrict__ qp,
       float* __restrict__ xenc, _Float16* __restrict__ xenc16,
       float* __restrict__ candR, float* __restrict__ candF) {
  __shared__ int      sTok[TTL];
  __shared__ __align__(16) _Float16 sEmb[32 * 320];   // K padded 300->320
  __shared__ __align__(16) _Float16 sQ[32 * 400];
  __shared__ __align__(16) _Float16 sK[32 * 400];
  __shared__ __align__(16) _Float16 sV[32 * 400];
  __shared__ __align__(16) _Float16 sOut[32 * DP];    // K padded 400->416
  __shared__ float    sLogit[32];
  __shared__ float    sA[32];

  int bid = blockIdx.x;
  int tid = threadIdx.x;
  const int* tok;
  if (bid < NN)                 tok = subx + (size_t)bid * TOK;
  else if (bid < NN + BB*NCAND) tok = cand + (size_t)(bid - NN) * TOK;
  else                          tok = fake + (size_t)(bid - NN - BB*NCAND) * TOK;

  if (tid < TTL) sTok[tid] = tok[tid];
  if (tid < 32)  sLogit[tid] = 0.f;
  __syncthreads();

  for (int i = tid; i < 32 * 320; i += 256) {
    int r = i / 320, c = i % 320;
    float v = 0.f;
    if (r < TTL && c < GD) v = glove[(size_t)sTok[r] * GD + c];
    sEmb[i] = (_Float16)v;
  }
  for (int i = tid; i < 32 * DP; i += 256) sOut[i] = (_Float16)0.f;
  __syncthreads();

  int wave = tid >> 5, lane = tid & 31;

  // ---- Q/K/V projections: 3 mats x 2 m-tiles x 25 n-tiles = 150 WMMA tiles ----
  for (int t = wave; t < 150; t += 8) {
    int w = t / 50, rem = t % 50, mt = rem / 25, nt = rem % 25;
    const unsigned int* W = (w == 0) ? wqP : (w == 1) ? wkP : wvP;
    _Float16* dst         = (w == 0) ? sQ  : (w == 1) ? sK  : sV;
    v8f c = {};
    #pragma unroll
    for (int kb = 0; kb < 10; ++kb) {          // Kpad = 320
      v16h a = load_a_p(sEmb, 320, mt * 16, kb * 32, lane);
      v16h b = load_b_p(W, 10, nt, kb, lane);
      c = wmma32(a, b, c);
    }
    int col = nt * 16 + (lane & 15);
    int rbase = mt * 16 + ((lane >> 4) ? 8 : 0);
    #pragma unroll
    for (int r = 0; r < 8; ++r) dst[(rbase + r) * 400 + col] = (_Float16)c[r];
  }
  __syncthreads();

  // ---- per-head attention (30 queries x 30 keys), VALU ----
  for (int h = wave; h < HH; h += 8) {
    if (lane < TTL) {
      int i = lane, base = h * DH;
      float qv[DH];
      #pragma unroll
      for (int d = 0; d < DH; ++d) qv[d] = (float)sQ[i * 400 + base + d];
      float s[TTL]; float mx = -1e30f;
      #pragma unroll
      for (int j = 0; j < TTL; ++j) {
        float acc = 0.f;
        #pragma unroll
        for (int d = 0; d < DH; ++d) acc += qv[d] * (float)sK[j * 400 + base + d];
        s[j] = acc * INVS;
        mx = fmaxf(mx, s[j]);
      }
      float sum = 0.f;
      #pragma unroll
      for (int j = 0; j < TTL; ++j) { s[j] = expf(s[j] - mx); sum += s[j]; }
      float o[DH];
      #pragma unroll
      for (int d = 0; d < DH; ++d) o[d] = 0.f;
      #pragma unroll
      for (int j = 0; j < TTL; ++j) {
        float w = s[j] / sum;
        #pragma unroll
        for (int d = 0; d < DH; ++d) o[d] += w * (float)sV[j * 400 + base + d];
      }
      #pragma unroll
      for (int d = 0; d < DH; ++d) sOut[i * DP + base + d] = (_Float16)o[d];
    }
  }
  __syncthreads();

  // ---- pooling: logits = tanh(out @ wp) . qp  (2 x 13 WMMA tiles, Kpad=416) ----
  for (int t = wave; t < 26; t += 8) {
    int mt = t / 13, nt = t % 13;
    v8f c = {};
    #pragma unroll
    for (int kb = 0; kb < 13; ++kb) {
      v16h a = load_a_p(sOut, DP, mt * 16, kb * 32, lane);
      v16h b = load_b_p(wpP, 13, nt, kb, lane);
      c = wmma32(a, b, c);
    }
    int col = nt * 16 + (lane & 15);
    int rbase = mt * 16 + ((lane >> 4) ? 8 : 0);
    if (col < 200) {
      float qpc = qp[col];
      #pragma unroll
      for (int r = 0; r < 8; ++r) {
        int row = rbase + r;
        if (row < TTL) atomicAdd(&sLogit[row], tanhf(c[r]) * qpc);  // ds_add_f32
      }
    }
  }
  __syncthreads();

  // ---- softmax over tokens (wave 0) ----
  if (wave == 0) {
    float l = (lane < TTL) ? sLogit[lane] : -1e30f;
    float m = l;
    #pragma unroll
    for (int off = 16; off; off >>= 1) m = fmaxf(m, __shfl_xor(m, off, 32));
    float e = (lane < TTL) ? expf(l - m) : 0.f;
    float s = e;
    #pragma unroll
    for (int off = 16; off; off >>= 1) s += __shfl_xor(s, off, 32);
    sA[lane] = e / s;
  }
  __syncthreads();

  // ---- weighted sum + store (f16 copy padded to DP) ----
  for (int d = tid; d < DP; d += 256) {
    float acc = 0.f;
    if (d < DD) {
      #pragma unroll 5
      for (int i = 0; i < TTL; ++i) acc += sA[i] * (float)sOut[i * DP + d];
    }
    if (bid < NN) {
      if (d < DD) xenc[(size_t)bid * DD + d] = acc;
      xenc16[(size_t)bid * DP + d] = (_Float16)acc;
    } else if (d < DD) {
      if (bid < NN + BB * NCAND) candR[(size_t)(bid - NN) * DD + d] = acc;
      else                       candF[(size_t)(bid - NN - BB * NCAND) * DD + d] = acc;
    }
  }
}

// ---------------- edge scatter-add: agg[dst] += m[src] ----------------
__global__ void __launch_bounds__(256)
k_scatter(const int* __restrict__ ei, const float* __restrict__ m,
          float* __restrict__ agg) {
  int e = blockIdx.x;
  int s = ei[e], dn = ei[EE + e];
  const float* mr = m + (size_t)s * DD;
  float* ar = agg + (size_t)dn * DD;
  for (int d = threadIdx.x; d < DD; d += 256)
    atomicAdd(&ar[d], mr[d]);
}

// ---------------- fused GRU update: 16 rows/block, gates via WMMA ----------------
__global__ void __launch_bounds__(256)
k_gru(const _Float16* __restrict__ agg16, const _Float16* __restrict__ h16,
      const float* __restrict__ hOld,
      const unsigned int* __restrict__ wihP, const unsigned int* __restrict__ whhP,
      const float* __restrict__ bih, const float* __restrict__ bhh,
      float* __restrict__ hNew, _Float16* __restrict__ hNew16) {
  __shared__ float sGi[16 * 1200];
  __shared__ float sGh[16 * 1200];
  int r0 = blockIdx.x * 16;
  int tid = threadIdx.x, wave = tid >> 5, lane = tid & 31;

  // 75 n-tiles for gi + 75 for gh ; K = 400 padded 416 -> KB = 13
  for (int t = wave; t < 150; t += 8) {
    bool isGh = t >= 75;
    int nt = isGh ? (t - 75) : t;
    const _Float16* A      = isGh ? h16  : agg16;
    const unsigned int* W  = isGh ? whhP : wihP;
    const float* bb        = isGh ? bhh  : bih;
    float* dst             = isGh ? sGh  : sGi;
    v8f c = {};
    #pragma unroll
    for (int kb = 0; kb < 13; ++kb) {
      v16h a = load_a_p(A, DP, r0, kb * 32, lane);
      v16h b = load_b_p(W, 13, nt, kb, lane);
      c = wmma32(a, b, c);
    }
    int col = nt * 16 + (lane & 15);
    int rbase = (lane >> 4) ? 8 : 0;
    float bv = bb[col];
    #pragma unroll
    for (int r = 0; r < 8; ++r)
      dst[(rbase + r) * 1200 + col] = c[r] + bv;
  }
  __syncthreads();

  for (int idx = tid; idx < 16 * DP; idx += 256) {
    int row = idx / DP, d = idx % DP;
    if (d < DD) {
      float i_r = sGi[row * 1200 + d];
      float i_z = sGi[row * 1200 + 400 + d];
      float i_n = sGi[row * 1200 + 800 + d];
      float h_r = sGh[row * 1200 + d];
      float h_z = sGh[row * 1200 + 400 + d];
      float h_n = sGh[row * 1200 + 800 + d];
      float r = 1.f / (1.f + expf(-(i_r + h_r)));
      float z = 1.f / (1.f + expf(-(i_z + h_z)));
      float n = tanhf(i_n + r * h_n);
      float hp = hOld[(size_t)(r0 + row) * DD + d];
      float hn = (1.f - z) * n + z * hp;
      hNew[(size_t)(r0 + row) * DD + d]   = hn;
      hNew16[(size_t)(r0 + row) * DP + d] = (_Float16)hn;
    } else {
      hNew16[(size_t)(r0 + row) * DP + d] = (_Float16)0.f;  // keep pad zero
    }
  }
}

// ---------------- gather + concat clicked views (f16 for GEMM A, K=800) ----------------
__global__ void k_gather(const int* __restrict__ mapping,
                         const float* __restrict__ xenc,
                         const float* __restrict__ hfin,
                         _Float16* __restrict__ ctin) {
  int total = BB * NC * 800;
  for (int i = blockIdx.x * blockDim.x + threadIdx.x; i < total;
       i += gridDim.x * blockDim.x) {
    int slot = i / 800, c = i % 800;
    int mi = mapping[slot];
    float v = 0.f;
    if (mi >= 0)
      v = (c < DD) ? xenc[(size_t)mi * DD + c] : hfin[(size_t)mi * DD + (c - DD)];
    ctin[i] = (_Float16)v;
  }
}

// ---------------- user MHSA (masked), B blocks ----------------
__global__ void __launch_bounds__(256)
k_uatt(const int* __restrict__ mapping, const float* __restrict__ Q,
       const float* __restrict__ Km, const float* __restrict__ Vm,
       float* __restrict__ Out, _Float16* __restrict__ Out16) {
  __shared__ float sBias[NC];
  int b = blockIdx.x, tid = threadIdx.x;
  if (tid < NC) sBias[tid] = (mapping[b * NC + tid] >= 0) ? 0.f : -1e9f;
  __syncthreads();
  for (int p = tid; p < NC * HH; p += 256) {
    int q = p / HH, h = p % HH;
    int rq = (b * NC + q) * DD + h * DH;
    float s[NC]; float mx = -1e30f;
    #pragma unroll
    for (int j = 0; j < NC; ++j) {
      int rk = (b * NC + j) * DD + h * DH;
      float acc = 0.f;
      #pragma unroll
      for (int d = 0; d < DH; ++d) acc += Q[rq + d] * Km[rk + d];
      s[j] = acc * INVS + sBias[j];
      mx = fmaxf(mx, s[j]);
    }
    float sum = 0.f;
    #pragma unroll
    for (int j = 0; j < NC; ++j) { s[j] = expf(s[j] - mx); sum += s[j]; }
    float o[DH];
    #pragma unroll
    for (int d = 0; d < DH; ++d) o[d] = 0.f;
    #pragma unroll
    for (int j = 0; j < NC; ++j) {
      float w = s[j] / sum;
      int rv = (b * NC + j) * DD + h * DH;
      #pragma unroll
      for (int d = 0; d < DH; ++d) o[d] += w * Vm[rv + d];
    }
    int rq16 = (b * NC + q) * DP + h * DH;
    #pragma unroll
    for (int d = 0; d < DH; ++d) {
      Out[rq + d] = o[d];
      Out16[rq16 + d] = (_Float16)o[d];
    }
  }
}

// ---------------- user attention pooling -> user_emb ----------------
__global__ void __launch_bounds__(256)
k_upool(const int* __restrict__ mapping, const float* __restrict__ P,
        const float* __restrict__ qp, const float* __restrict__ Out,
        float* __restrict__ uemb) {
  __shared__ float sL[NC], sPA[NC];
  int b = blockIdx.x, tid = threadIdx.x;
  for (int q = tid; q < NC; q += 256) {
    float acc = 0.f;
    const float* pr = P + (size_t)(b * NC + q) * 200;
    for (int c = 0; c < 200; ++c) acc += tanhf(pr[c]) * qp[c];
    sL[q] = acc + ((mapping[b * NC + q] >= 0) ? 0.f : -1e9f);
  }
  __syncthreads();
  if (tid == 0) {
    float mx = -1e30f;
    for (int q = 0; q < NC; ++q) mx = fmaxf(mx, sL[q]);
    float sum = 0.f;
    for (int q = 0; q < NC; ++q) { float e = expf(sL[q] - mx); sPA[q] = e; sum += e; }
    float inv = 1.f / sum;
    for (int q = 0; q < NC; ++q) sPA[q] *= inv;
  }
  __syncthreads();
  for (int d = tid; d < DD; d += 256) {
    float acc = 0.f;
    for (int q = 0; q < NC; ++q) acc += sPA[q] * Out[(size_t)(b * NC + q) * DD + d];
    uemb[b * DD + d] = acc;
  }
}

// ---------------- candidate dot-product scores ----------------
__global__ void __launch_bounds__(128)
k_score(const float* __restrict__ candR, const float* __restrict__ candF,
        const float* __restrict__ uemb, float* __restrict__ out) {
  int idx = blockIdx.x;                // 0..639
  bool fk = idx >= BB * NCAND;
  int ci = fk ? idx - BB * NCAND : idx;
  int b  = ci / NCAND;
  const float* c = (fk ? candF : candR) + (size_t)ci * DD;
  const float* u = uemb + b * DD;
  float acc = 0.f;
  for (int d = threadIdx.x; d < DD; d += 128) acc += c[d] * u[d];
  #pragma unroll
  for (int off = 16; off; off >>= 1) acc += __shfl_xor(acc, off, 32);
  __shared__ float sW[4];
  if ((threadIdx.x & 31) == 0) sW[threadIdx.x >> 5] = acc;
  __syncthreads();
  if (threadIdx.x == 0)
    out[(fk ? 1 + BB * NCAND : 1) + ci] = sW[0] + sW[1] + sW[2] + sW[3];
}

// ---------------- NCE + uniformity loss ----------------
__global__ void k_loss(const int* __restrict__ label, float* __restrict__ out) {
  __shared__ float sR[BB], sF[BB];
  int b = threadIdx.x;                 // block of 64
  const float* sr = out + 1 + b * NCAND;
  const float* sf = out + 1 + BB * NCAND + b * NCAND;
  float m = -1e30f;
  for (int c = 0; c < NCAND; ++c) m = fmaxf(m, sr[c]);
  float s = 0.f;
  for (int c = 0; c < NCAND; ++c) s += expf(sr[c] - m);
  float lse = logf(s) + m;
  sR[b] = -(sr[label[b]] - lse);
  float mf = -1e30f;
  for (int c = 0; c < NCAND; ++c) mf = fmaxf(mf, sf[c]);
  float sfx = 0.f;
  for (int c = 0; c < NCAND; ++c) sfx += expf(sf[c] - mf);
  float lsef = logf(sfx) + mf;
  float acc = 0.f;
  for (int c = 0; c < NCAND; ++c) acc += -(sf[c] - lsef);
  sF[b] = acc;
  __syncthreads();
  if (b == 0) {
    float r = 0.f, f = 0.f;
    for (int i = 0; i < BB; ++i) { r += sR[i]; f += sF[i]; }
    out[0] = r / BB + 0.5f * (f / (BB * NCAND));
  }
}

// ---------------- host driver ----------------
extern "C" void kernel_launch(void* const* d_in, const int* in_sizes, int n_in,
                              void* d_out, int out_size, void* d_ws, size_t ws_size,
                              hipStream_t stream) {
  const int*   subx    = (const int*)d_in[0];
  const int*   ei      = (const int*)d_in[1];
  const int*   mapping = (const int*)d_in[2];
  const int*   cand    = (const int*)d_in[3];
  const int*   fake    = (const int*)d_in[4];
  const int*   label   = (const int*)d_in[5];
  const float* glove   = (const float*)d_in[6];
  const float* n_wq    = (const float*)d_in[7];
  const float* n_wk    = (const float*)d_in[8];
  const float* n_wv    = (const float*)d_in[9];
  const float* n_wp    = (const float*)d_in[10];
  const float* n_qp    = (const float*)d_in[11];
  const float* ggc_w   = (const float*)d_in[12];
  const float* gru_wih = (const float*)d_in[13];
  const float* gru_whh = (const float*)d_in[14];
  const float* gru_bih = (const float*)d_in[15];
  const float* gru_bhh = (const float*)d_in[16];
  const float* w_click = (const float*)d_in[17];
  const float* b_click = (const float*)d_in[18];
  const float* u_wq    = (const float*)d_in[19];
  const float* u_wk    = (const float*)d_in[20];
  const float* u_wv    = (const float*)d_in[21];
  const float* u_wp    = (const float*)d_in[22];
  const float* u_qp    = (const float*)d_in[23];
  float* out = (float*)d_out;
  char*  ws  = (char*)d_ws;

  size_t off = 0;
  auto alloc = [&](size_t bytes) -> char* {
    char* p = ws + off;
    off = (off + bytes + 255) & ~(size_t)255;
    return p;
  };
  // packed weights (u32 words = NT*KB*256)
  auto packBytes = [](int KB, int NT) { return (size_t)NT * KB * 256 * 4; };
  unsigned int* wqP  = (unsigned int*)alloc(packBytes(10, 25));   // 300->320, N=400
  unsigned int* wkP  = (unsigned int*)alloc(packBytes(10, 25));
  unsigned int* wvP  = (unsigned int*)alloc(packBytes(10, 25));
  unsigned int* wpP  = (unsigned int*)alloc(packBytes(13, 13));   // 400->416, N=200
  unsigned int* ggcP = (unsigned int*)alloc(3 * packBytes(13, 25));
  unsigned int* wihP = (unsigned int*)alloc(packBytes(13, 75));
  unsigned int* whhP = (unsigned int*)alloc(packBytes(13, 75));
  unsigned int* wclP = (unsigned int*)alloc(packBytes(25, 25));   // K=800, N=400
  unsigned int* uwqP = (unsigned int*)alloc(packBytes(13, 25));
  unsigned int* uwkP = (unsigned int*)alloc(packBytes(13, 25));
  unsigned int* uwvP = (unsigned int*)alloc(packBytes(13, 25));
  unsigned int* uwpP = (unsigned int*)alloc(packBytes(13, 13));
  // activations
  float*    xenc   = (float*)   alloc((size_t)NN * DD * 4);
  _Float16* xenc16 = (_Float16*)alloc((size_t)NN * DP * 2);
  float*    h1     = (float*)   alloc((size_t)NN * DD * 4);
  _Float16* h1_16  = (_Float16*)alloc((size_t)NN * DP * 2);
  float*    h2     = (float*)   alloc((size_t)NN * DD * 4);
  _Float16* h2_16  = (_Float16*)alloc((size_t)NN * DP * 2);
  float*    mbuf   = (float*)   alloc((size_t)NN * DD * 4);
  float*    agg    = (float*)   alloc((size_t)NN * DD * 4);
  _Float16* agg16  = (_Float16*)alloc((size_t)NN * DP * 2);
  float*    candR  = (float*)   alloc((size_t)BB * NCAND * DD * 4);
  float*    candF  = (float*)   alloc((size_t)BB * NCAND * DD * 4);
  _Float16* ctin16 = (_Float16*)alloc((size_t)BB * NC * 800 * 2);
  float*    ct     = (float*)   alloc((size_t)BB * NC * DD * 4);
  _Float16* ct16   = (_Float16*)alloc((size_t)BB * NC * DP * 2);
  float*    uq     = (float*)   alloc((size_t)BB * NC * DD * 4);
  float*    uk     = (float*)   alloc((size_t)BB * NC * DD * 4);
  float*    uv     = (float*)   alloc((size_t)BB * NC * DD * 4);
  float*    uout   = (float*)   alloc((size_t)BB * NC * DD * 4);
  _Float16* uout16 = (_Float16*)alloc((size_t)BB * NC * DP * 2);
  float*    Pbuf   = (float*)   alloc((size_t)BB * NC * 200 * 4);
  float*    uemb   = (float*)   alloc((size_t)BB * DD * 4);
  (void)ws_size; (void)in_sizes; (void)n_in; (void)out_size;

  auto pack = [&](const float* s, unsigned int* d, int K, int N, int KB, int NT) {
    int total = NT * KB * 256;
    k_packB<<<(total + 255) / 256, 256, 0, stream>>>(s, d, K, N, KB, NT);
  };
  pack(n_wq, wqP, GD, DD, 10, 25);
  pack(n_wk, wkP, GD, DD, 10, 25);
  pack(n_wv, wvP, GD, DD, 10, 25);
  pack(n_wp, wpP, DD, 200, 13, 13);
  for (int l = 0; l < 3; ++l)
    pack(ggc_w + (size_t)l * DD * DD, ggcP + (size_t)l * 13 * 25 * 256, DD, DD, 13, 25);
  pack(gru_wih, wihP, DD, 1200, 13, 75);
  pack(gru_whh, whhP, DD, 1200, 13, 75);
  pack(w_click, wclP, 800, DD, 25, 25);
  pack(u_wq, uwqP, DD, DD, 13, 25);
  pack(u_wk, uwkP, DD, DD, 13, 25);
  pack(u_wv, uwvP, DD, DD, 13, 25);
  pack(u_wp, uwpP, DD, 200, 13, 13);

  // news encoder over all nodes + real + fake candidates
  k_news<<<NN + 2 * BB * NCAND, 256, 0, stream>>>(
      subx, cand, fake, glove, wqP, wkP, wvP, wpP, n_qp,
      xenc, xenc16, candR, candF);

  auto gemmBlocks = [](int M, int N) {
    int tiles = ((M + 15) / 16) * ((N + 15) / 16);
    return (tiles * 32 + 255) / 256;
  };

  // ---- 3 GatedGraphConv layers ----
  const float*    hF[4]  = {xenc, h1, h2, h1};
  const _Float16* hH[4]  = {xenc16, h1_16, h2_16, h1_16};
  float*          hNf[3] = {h1, h2, h1};
  _Float16*       hNh[3] = {h1_16, h2_16, h1_16};
  for (int l = 0; l < 3; ++l) {
    k_gemm<<<gemmBlocks(NN, DD), 256, 0, stream>>>(
        hH[l], DP, ggcP + (size_t)l * 13 * 25 * 256, nullptr, mbuf,
        (_Float16*)nullptr, 0, NN, 13, DD);
    k_zero<<<(NN * DD + 255) / 256, 256, 0, stream>>>(agg, NN * DD);
    k_scatter<<<EE, 256, 0, stream>>>(ei, mbuf, agg);
    k_f32tof16_pad<<<(NN * DP + 255) / 256, 256, 0, stream>>>(agg, agg16, NN, DD, DP);
    k_gru<<<NN / 16, 256, 0, stream>>>(agg16, hH[l], hF[l], wihP, whhP,
                                       gru_bih, gru_bhh, hNf[l], hNh[l]);
  }
  const float* hfin = hF[3];

  // ---- click encoder ----
  k_gather<<<(BB * NC * 800 + 255) / 256, 256, 0, stream>>>(mapping, xenc, hfin,
                                                            ctin16);
  k_zero16<<<(BB * NC * DP + 255) / 256, 256, 0, stream>>>(ct16, BB * NC * DP);
  k_gemm<<<gemmBlocks(BB * NC, DD), 256, 0, stream>>>(
      ctin16, 800, wclP, b_click, ct, ct16, DP, BB * NC, 25, DD);

  // ---- user encoder ----
  k_gemm<<<gemmBlocks(BB * NC, DD), 256, 0, stream>>>(
      ct16, DP, uwqP, nullptr, uq, (_Float16*)nullptr, 0, BB * NC, 13, DD);
  k_gemm<<<gemmBlocks(BB * NC, DD), 256, 0, stream>>>(
      ct16, DP, uwkP, nullptr, uk, (_Float16*)nullptr, 0, BB * NC, 13, DD);
  k_gemm<<<gemmBlocks(BB * NC, DD), 256, 0, stream>>>(
      ct16, DP, uwvP, nullptr, uv, (_Float16*)nullptr, 0, BB * NC, 13, DD);
  k_zero16<<<(BB * NC * DP + 255) / 256, 256, 0, stream>>>(uout16, BB * NC * DP);
  k_uatt<<<BB, 256, 0, stream>>>(mapping, uq, uk, uv, uout, uout16);
  k_gemm<<<gemmBlocks(BB * NC, 200), 256, 0, stream>>>(
      uout16, DP, uwpP, nullptr, Pbuf, (_Float16*)nullptr, 0, BB * NC, 13, 200);
  k_upool<<<BB, 256, 0, stream>>>(mapping, Pbuf, u_qp, uout, uemb);

  // ---- scoring + loss ----
  k_score<<<2 * BB * NCAND, 128, 0, stream>>>(candR, candF, uemb, out);
  k_loss<<<1, BB, 0, stream>>>(label, out);
}
